// GATLP_23965917511987
// MI455X (gfx1250) — matrix operand order repaired and checked by reading, not compiled, not run on hardware
//
#include <hip/hip_runtime.h>
#include <hip/hip_bf16.h>

// ============================================================================
// 2-layer GAT (PyG-style) for MI455X / gfx1250.
//  - Dense transforms x@W1, h@W2 via v_wmma_f32_16x16x32_bf16 (f32 accumulate),
//    B staged in LDS as bf16 (transposed, padded), K-loop fully unrolled.
//  - Edge softmax + aggregation via fp32 VALU + float atomics (L2-resident)
// ============================================================================

#define NEG_SLOPE 0.2f

typedef __attribute__((ext_vector_type(16))) __bf16 v16bf;
typedef __attribute__((ext_vector_type(8)))  float  v8f;

__device__ __forceinline__ float lrelu(float v) {
    return v > 0.0f ? v : NEG_SLOPE * v;
}

// ---------------------------------------------------------------------------
// fill
// ---------------------------------------------------------------------------
__global__ void fill_f32(float* __restrict__ p, int n, float v) {
    int i = blockIdx.x * blockDim.x + threadIdx.x;
    if (i < n) p[i] = v;
}

// ---------------------------------------------------------------------------
// C[M x 64] = A[M x K](f32) * B[K x 64](f32), bf16 WMMA, f32 accumulate.
// 4 waves / block; wave w owns columns [16w, 16w+16); block owns 16 rows.
// B is staged once per block into LDS, transposed (n-major) and converted to
// bf16 with padded stride KP so each lane's 16-element K-run is a pair of
// 16B-aligned contiguous chunks (-> ds_load_b128).
//
// WMMA operand layouts used here (A/B contiguous k, this code's choice; D per ISA 7.12.2):
//  A (16-bit, 16x32): lane l -> row l%16;
//     elem i = K (l<16?0:16)+i (same k order as B)
//  B (32x16): lane l -> col l%16; elem i = K (l<16?0:16)+i
//  D: vgpr v -> row v + (l<16?0:8), col l%16
// ---------------------------------------------------------------------------
template <int K>
__global__ __launch_bounds__(128)
void gemm_bf16_wmma(const float* __restrict__ A, const float* __restrict__ B,
                    float* __restrict__ C, int M) {
    constexpr int NOUT = 64;
    constexpr int KP   = K + 8;                 // bf16 elems; multiple of 8
    __shared__ alignas(16) __bf16 Bs[NOUT * KP];

    // ---- cooperative transpose-stage of W into LDS (bf16) ----
    for (int idx = threadIdx.x; idx < K * NOUT; idx += 128) {
        const int k = idx >> 6;                 // row of W
        const int n = idx & 63;                 // col of W
        Bs[n * KP + k] = (__bf16)B[idx];
    }
    __syncthreads();

    const int lane = threadIdx.x & 31;
    const int wave = threadIdx.x >> 5;          // 0..3 -> n-tile
    const int l16  = lane & 15;
    const int half = lane >> 4;                 // 0 or 1
    const int m0   = blockIdx.x * 16;

    int arow = m0 + l16;
    if (arow >= M) arow = M - 1;                // clamp: keep EXEC all-ones
    const float* __restrict__ ap = A + (size_t)arow * K;
    const int ncol = wave * 16 + l16;
    const __bf16* __restrict__ bp = &Bs[ncol * KP];

    v8f acc = {};
#pragma unroll
    for (int k0 = 0; k0 < K; k0 += 32) {
        v16bf av, bv;
        const int ka0 = k0 + (half ? 16 : 0);
        const int ka1 = k0 + (half ? 24 : 8);
#pragma unroll
        for (int i = 0; i < 8; ++i) {
            av[i]     = (__bf16)ap[ka0 + i];
            av[8 + i] = (__bf16)ap[ka1 + i];
        }
        const __bf16* __restrict__ q = bp + k0 + (half ? 16 : 0);
#pragma unroll
        for (int i = 0; i < 16; ++i) bv[i] = q[i];

        acc = __builtin_amdgcn_wmma_f32_16x16x32_bf16(
            false, av, false, bv, (short)0, acc, false, false);
    }

    if (m0 + 16 <= M) {                         // whole-tile fast path
#pragma unroll
        for (int v = 0; v < 8; ++v)
            C[(size_t)(m0 + v + half * 8) * NOUT + ncol] = acc[v];
    } else {                                    // tail tile (never hit: M%16==0)
#pragma unroll
        for (int v = 0; v < 8; ++v) {
            const int m = m0 + v + half * 8;
            if (m < M) C[(size_t)m * NOUT + ncol] = acc[v];
        }
    }
}

// ---------------------------------------------------------------------------
// Attention projections
// ---------------------------------------------------------------------------
// layer 1: 8 heads x 8 ch; one thread per (node, head)
__global__ void att_heads8(const float* __restrict__ h,
                           const float* __restrict__ asrc,
                           const float* __restrict__ adst,
                           float* __restrict__ os, float* __restrict__ od,
                           int n_nodes) {
    int tid = blockIdx.x * blockDim.x + threadIdx.x;
    if (tid >= n_nodes * 8) return;
    const int hd = tid & 7;
    const int node = tid >> 3;
    const float* __restrict__ hp = h + (size_t)node * 64 + hd * 8;
    const float* __restrict__ ws = asrc + hd * 8;
    const float* __restrict__ wd = adst + hd * 8;
    float ss = 0.f, dd = 0.f;
#pragma unroll
    for (int c = 0; c < 8; ++c) { ss += hp[c] * ws[c]; dd += hp[c] * wd[c]; }
    os[tid] = ss;
    od[tid] = dd;
}

// layer 2: 1 head x 64 ch; one thread per node
__global__ void att_head1(const float* __restrict__ h,
                          const float* __restrict__ asrc,
                          const float* __restrict__ adst,
                          float* __restrict__ os, float* __restrict__ od,
                          int n_nodes) {
    int node = blockIdx.x * blockDim.x + threadIdx.x;
    if (node >= n_nodes) return;
    const float* __restrict__ hp = h + (size_t)node * 64;
    float ss = 0.f, dd = 0.f;
#pragma unroll
    for (int j = 0; j < 64; ++j) { ss += hp[j] * asrc[j]; dd += hp[j] * adst[j]; }
    os[node] = ss;
    od[node] = dd;
}

// ---------------------------------------------------------------------------
// Edge helpers: edges 0..E-1 from edge_index, edges E..E+N-1 are self-loops
// ---------------------------------------------------------------------------
__device__ __forceinline__ void edge_sd(const int* __restrict__ ei, int E,
                                        int e, int& s, int& d) {
    if (e < E) { s = ei[e]; d = ei[E + e]; }
    else       { s = d = e - E; }
}

// ---- layer 1 (H=8): one thread per (edge, head) ----
__global__ void edge_max_h8(const int* __restrict__ ei,
                            const float* __restrict__ as,
                            const float* __restrict__ ad,
                            float* __restrict__ m, int E, int n_nodes) {
    int tid = blockIdx.x * blockDim.x + threadIdx.x;
    if (tid >= (E + n_nodes) * 8) return;
    const int hd = tid & 7, e = tid >> 3;
    int s, d; edge_sd(ei, E, e, s, d);
    const float v = lrelu(as[(size_t)s * 8 + hd] + ad[(size_t)d * 8 + hd]);
    atomicMax(&m[(size_t)d * 8 + hd], v);   // global_atomic_max_num_f32
}

__global__ void edge_expsum_h8(const int* __restrict__ ei,
                               const float* __restrict__ as,
                               const float* __restrict__ ad,
                               const float* __restrict__ m,
                               float* __restrict__ ssum, int E, int n_nodes) {
    int tid = blockIdx.x * blockDim.x + threadIdx.x;
    if (tid >= (E + n_nodes) * 8) return;
    const int hd = tid & 7, e = tid >> 3;
    int s, d; edge_sd(ei, E, e, s, d);
    const size_t di = (size_t)d * 8 + hd;
    const float v = lrelu(as[(size_t)s * 8 + hd] + ad[di]);
    atomicAdd(&ssum[di], __expf(v - m[di]));
}

__global__ void edge_agg_h8(const int* __restrict__ ei,
                            const float* __restrict__ as,
                            const float* __restrict__ ad,
                            const float* __restrict__ m,
                            const float* __restrict__ ssum,
                            const float* __restrict__ h,
                            float* __restrict__ agg, int E, int n_nodes) {
    int tid = blockIdx.x * blockDim.x + threadIdx.x;
    if (tid >= (E + n_nodes) * 8) return;
    const int hd = tid & 7, e = tid >> 3;
    int s, d; edge_sd(ei, E, e, s, d);
    const size_t di = (size_t)d * 8 + hd;
    const float v = lrelu(as[(size_t)s * 8 + hd] + ad[di]);
    const float alpha = __expf(v - m[di]) / ssum[di];
    const float4* __restrict__ hp =
        (const float4*)(h + (size_t)s * 64 + hd * 8);
    const float4 a0 = hp[0], a1 = hp[1];
    float* __restrict__ op = agg + (size_t)d * 64 + hd * 8;
    atomicAdd(op + 0, alpha * a0.x);
    atomicAdd(op + 1, alpha * a0.y);
    atomicAdd(op + 2, alpha * a0.z);
    atomicAdd(op + 3, alpha * a0.w);
    atomicAdd(op + 4, alpha * a1.x);
    atomicAdd(op + 5, alpha * a1.y);
    atomicAdd(op + 6, alpha * a1.z);
    atomicAdd(op + 7, alpha * a1.w);
}

// ---- layer 2 (H=1): one thread per edge (max/sum), 8 threads/edge (agg) ----
__global__ void edge_max_h1(const int* __restrict__ ei,
                            const float* __restrict__ as,
                            const float* __restrict__ ad,
                            float* __restrict__ m, int E, int n_nodes) {
    int e = blockIdx.x * blockDim.x + threadIdx.x;
    if (e >= E + n_nodes) return;
    int s, d; edge_sd(ei, E, e, s, d);
    atomicMax(&m[d], lrelu(as[s] + ad[d]));
}

__global__ void edge_expsum_h1(const int* __restrict__ ei,
                               const float* __restrict__ as,
                               const float* __restrict__ ad,
                               const float* __restrict__ m,
                               float* __restrict__ ssum, int E, int n_nodes) {
    int e = blockIdx.x * blockDim.x + threadIdx.x;
    if (e >= E + n_nodes) return;
    int s, d; edge_sd(ei, E, e, s, d);
    const float v = lrelu(as[s] + ad[d]);
    atomicAdd(&ssum[d], __expf(v - m[d]));
}

__global__ void edge_agg_c64(const int* __restrict__ ei,
                             const float* __restrict__ as,
                             const float* __restrict__ ad,
                             const float* __restrict__ m,
                             const float* __restrict__ ssum,
                             const float* __restrict__ h,
                             float* __restrict__ agg, int E, int n_nodes) {
    int tid = blockIdx.x * blockDim.x + threadIdx.x;
    if (tid >= (E + n_nodes) * 8) return;
    const int part = tid & 7, e = tid >> 3;
    int s, d; edge_sd(ei, E, e, s, d);
    const float v = lrelu(as[s] + ad[d]);
    const float alpha = __expf(v - m[d]) / ssum[d];
    const float4* __restrict__ hp =
        (const float4*)(h + (size_t)s * 64 + part * 8);
    const float4 a0 = hp[0], a1 = hp[1];
    float* __restrict__ op = agg + (size_t)d * 64 + part * 8;
    atomicAdd(op + 0, alpha * a0.x);
    atomicAdd(op + 1, alpha * a0.y);
    atomicAdd(op + 2, alpha * a0.z);
    atomicAdd(op + 3, alpha * a0.w);
    atomicAdd(op + 4, alpha * a1.x);
    atomicAdd(op + 5, alpha * a1.y);
    atomicAdd(op + 6, alpha * a1.z);
    atomicAdd(op + 7, alpha * a1.w);
}

// ---------------------------------------------------------------------------
// epilogues
// ---------------------------------------------------------------------------
__global__ void relu_bias64(float* __restrict__ h, const float* __restrict__ b,
                            int n_nodes) {
    int tid = blockIdx.x * blockDim.x + threadIdx.x;
    if (tid >= n_nodes * 64) return;
    h[tid] = fmaxf(h[tid] + b[tid & 63], 0.0f);
}

__global__ void bias_logsoftmax64(const float* __restrict__ agg,
                                  const float* __restrict__ b,
                                  float* __restrict__ out, int n_nodes) {
    int node = blockIdx.x * blockDim.x + threadIdx.x;
    if (node >= n_nodes) return;
    const float* __restrict__ ip = agg + (size_t)node * 64;
    float v[64];
    float mx = -3.0e38f;
#pragma unroll
    for (int j = 0; j < 64; ++j) { v[j] = ip[j] + b[j]; mx = fmaxf(mx, v[j]); }
    float sum = 0.f;
#pragma unroll
    for (int j = 0; j < 64; ++j) sum += __expf(v[j] - mx);
    const float lse = __logf(sum) + mx;
    float* __restrict__ op = out + (size_t)node * 64;
#pragma unroll
    for (int j = 0; j < 64; ++j) op[j] = v[j] - lse;
}

// ---------------------------------------------------------------------------
// launch
// ---------------------------------------------------------------------------
static inline unsigned nblk(long long t) { return (unsigned)((t + 255) / 256); }

extern "C" void kernel_launch(void* const* d_in, const int* in_sizes, int n_in,
                              void* d_out, int out_size, void* d_ws, size_t ws_size,
                              hipStream_t stream) {
    (void)n_in; (void)out_size; (void)ws_size;
    const float* x    = (const float*)d_in[0];
    const int*   ei   = (const int*)d_in[1];        // [2,E] flat: src row, dst row
    const float* W1   = (const float*)d_in[2];
    const float* a_s1 = (const float*)d_in[3];
    const float* a_d1 = (const float*)d_in[4];
    const float* b1   = (const float*)d_in[5];
    const float* W2   = (const float*)d_in[6];
    const float* a_s2 = (const float*)d_in[7];
    const float* a_d2 = (const float*)d_in[8];
    const float* b2   = (const float*)d_in[9];
    float* out = (float*)d_out;

    const int N = in_sizes[0] / 128;
    const int E = in_sizes[1] / 2;
    const int EN = E + N;

    // workspace carve-up (fp32); total = 292*N floats (~117 MB for N=100k)
    float* w = (float*)d_ws;
    size_t o = 0;
    float* h1    = w + o; o += (size_t)N * 64;
    float* as1   = w + o; o += (size_t)N * 8;
    float* ad1   = w + o; o += (size_t)N * 8;
    float* m1    = w + o; o += (size_t)N * 8;
    float* sum1  = w + o; o += (size_t)N * 8;
    float* agg1  = w + o; o += (size_t)N * 64;   // becomes relu(h) in-place
    float* h2    = w + o; o += (size_t)N * 64;
    float* as2   = w + o; o += (size_t)N;
    float* ad2   = w + o; o += (size_t)N;
    float* m2    = w + o; o += (size_t)N;
    float* sum2  = w + o; o += (size_t)N;
    float* agg2  = w + o; o += (size_t)N * 64;

    const int TB = 256;

    // deterministic re-init each call (harness does not re-poison between replays)
    fill_f32<<<nblk((long long)N * 8),  TB, 0, stream>>>(m1,   N * 8, -3.0e38f);
    fill_f32<<<nblk((long long)N * 8),  TB, 0, stream>>>(sum1, N * 8, 0.0f);
    fill_f32<<<nblk((long long)N * 64), TB, 0, stream>>>(agg1, N * 64, 0.0f);
    fill_f32<<<nblk(N),                 TB, 0, stream>>>(m2,   N, -3.0e38f);
    fill_f32<<<nblk(N),                 TB, 0, stream>>>(sum2, N, 0.0f);
    fill_f32<<<nblk((long long)N * 64), TB, 0, stream>>>(agg2, N * 64, 0.0f);

    // ---- layer 1: 8 heads x 8 ch ----
    gemm_bf16_wmma<128><<<(N + 15) / 16, 128, 0, stream>>>(x, W1, h1, N);
    att_heads8<<<nblk((long long)N * 8), TB, 0, stream>>>(h1, a_s1, a_d1, as1, ad1, N);
    edge_max_h8   <<<nblk((long long)EN * 8), TB, 0, stream>>>(ei, as1, ad1, m1, E, N);
    edge_expsum_h8<<<nblk((long long)EN * 8), TB, 0, stream>>>(ei, as1, ad1, m1, sum1, E, N);
    edge_agg_h8   <<<nblk((long long)EN * 8), TB, 0, stream>>>(ei, as1, ad1, m1, sum1, h1, agg1, E, N);
    relu_bias64<<<nblk((long long)N * 64), TB, 0, stream>>>(agg1, b1, N);

    // ---- layer 2: 1 head x 64 ch ----
    gemm_bf16_wmma<64><<<(N + 15) / 16, 128, 0, stream>>>(agg1, W2, h2, N);
    att_head1<<<nblk(N), TB, 0, stream>>>(h2, a_s2, a_d2, as2, ad2, N);
    edge_max_h1   <<<nblk(EN), TB, 0, stream>>>(ei, as2, ad2, m2, E, N);
    edge_expsum_h1<<<nblk(EN), TB, 0, stream>>>(ei, as2, ad2, m2, sum2, E, N);
    edge_agg_c64  <<<nblk((long long)EN * 8), TB, 0, stream>>>(ei, as2, ad2, m2, sum2, h2, agg2, E, N);

    // ---- bias + log_softmax ----
    bias_logsoftmax64<<<nblk(N), TB, 0, stream>>>(agg2, b2, out, N);
}
